// Decoder_14602888807044
// MI455X (gfx1250) — matrix operand order repaired and checked by reading, not compiled, not run on hardware
//
#include <hip/hip_runtime.h>

// ---------------------------------------------------------------------------
// CDNA5 (gfx1250) persistent GRU-decoder.
//  - all matmuls as v_wmma_f32_16x16x32_bf16 (M=16 == batch, one wave per tile)
//  - weights/embedding converted fp32->bf16 once per call (L2-resident after)
//  - one persistent kernel; 6 device-wide barriers per timestep
// ---------------------------------------------------------------------------

typedef __attribute__((ext_vector_type(16))) __bf16 v16bf;
typedef __attribute__((ext_vector_type(8)))  __bf16 v8bf;
typedef __attribute__((ext_vector_type(8)))  float  v8f;

#define KV 32000
#define KE 512
#define KH 1024
#define KB 16
#define KS 128
#define K3 3072

#define PGRID  96
#define PBLOCK 256
#define NWAVES (PGRID * (PBLOCK / 32))

// ---------------- helpers ----------------

__device__ __forceinline__ unsigned short f2bf(float f) {
  unsigned u = __float_as_uint(f);
  unsigned r = 0x7FFFu + ((u >> 16) & 1u);   // round-to-nearest-even
  return (unsigned short)((u + r) >> 16);
}

__device__ __forceinline__ float sigm(float x) {
  return 1.f / (1.f + __expf(-x));
}

// One wave computes a 16x16 f32 tile of  A(16xK) @ B(KxN).T-slice.
// arow: this lane's A row (row index = lane&15), bf16, contiguous in k.
// bmat: weights, row n contiguous in k, leading dim ldb.
__device__ __forceinline__ v8f tile_gemm_bf16(
    const unsigned short* __restrict__ arow,
    const unsigned short* __restrict__ bmat,
    int ldb, int n0, int K, int lane) {
  const int half = lane >> 4;                                  // K-half select
  const unsigned short* ap = arow + (half << 4);               // +0 / +16
  const unsigned short* bp = bmat + (size_t)(n0 + (lane & 15)) * ldb + (half << 4);
  v8f acc = {0.f, 0.f, 0.f, 0.f, 0.f, 0.f, 0.f, 0.f};
  for (int k0 = 0; k0 < K; k0 += 32) {
    __builtin_prefetch((const void*)(bp + k0 + 128), 0, 0);    // global_prefetch_b8
    v8bf a0 = *(const v8bf*)(const void*)(ap + k0);            // K base..base+7
    v8bf a1 = *(const v8bf*)(const void*)(ap + k0 + 8);        // K base+8..+15
    v16bf av = __builtin_shufflevector(a0, a1,
        0, 1, 2, 3, 4, 5, 6, 7, 8, 9, 10, 11, 12, 13, 14, 15);
    v16bf bv = *(const v16bf*)(const void*)(bp + k0);          // 16 contiguous K
    acc = __builtin_amdgcn_wmma_f32_16x16x32_bf16(
        false, av, false, bv, (short)0, acc, false, false);
  }
  return acc;
}

// device-wide sense-reversing barrier (agent scope, spins in L2)
__device__ __forceinline__ void grid_sync(unsigned* cnt, unsigned* gen) {
  __syncthreads();
  if (threadIdx.x == 0) {
    unsigned g = __hip_atomic_load(gen, __ATOMIC_RELAXED, __HIP_MEMORY_SCOPE_AGENT);
    __threadfence();
    unsigned a = __hip_atomic_fetch_add(cnt, 1u, __ATOMIC_ACQ_REL,
                                        __HIP_MEMORY_SCOPE_AGENT);
    if (a == (unsigned)(PGRID - 1)) {
      __hip_atomic_store(cnt, 0u, __ATOMIC_RELAXED, __HIP_MEMORY_SCOPE_AGENT);
      __hip_atomic_fetch_add(gen, 1u, __ATOMIC_RELEASE, __HIP_MEMORY_SCOPE_AGENT);
    } else {
      while (__hip_atomic_load(gen, __ATOMIC_ACQUIRE,
                               __HIP_MEMORY_SCOPE_AGENT) == g) {
        __builtin_amdgcn_s_sleep(2);
      }
    }
  }
  __syncthreads();
}

// ---------------- one-time prep kernels ----------------

__global__ void cvt_f32_bf16(const float* __restrict__ src,
                             unsigned short* __restrict__ dst, size_t n) {
  size_t i = (size_t)blockIdx.x * blockDim.x + threadIdx.x;
  size_t st = (size_t)gridDim.x * blockDim.x;
  for (; i < n; i += st) dst[i] = f2bf(src[i]);
}

__global__ void init_state(const float* __restrict__ hidden,
                           float* h0f, unsigned short* h0b,
                           float* h1f, unsigned short* h1b, unsigned* bar) {
  int i = blockIdx.x * blockDim.x + threadIdx.x;
  if (i == 0) { bar[0] = 0u; bar[1] = 0u; }
  if (i < KB * KH) {
    float a = hidden[i];
    float b = hidden[KB * KH + i];
    h0f[i] = a; h0b[i] = f2bf(a);
    h1f[i] = b; h1b[i] = f2bf(b);
  }
}

// ---------------- persistent recurrence kernel ----------------

__global__ __launch_bounds__(PBLOCK) void decoder_persistent(
    const int* __restrict__ trg,
    const unsigned short* __restrict__ embbf,
    const unsigned short* __restrict__ wih0, const unsigned short* __restrict__ whh0,
    const float* __restrict__ bih0, const float* __restrict__ bhh0,
    const unsigned short* __restrict__ wih1, const unsigned short* __restrict__ whh1,
    const float* __restrict__ bih1, const float* __restrict__ bhh1,
    const unsigned short* __restrict__ w1b, const float* __restrict__ b1,
    const unsigned short* __restrict__ w2b, const float* __restrict__ b2,
    const float* __restrict__ bgen,
    float* h0f, unsigned short* h0b, float* h1f, unsigned short* h1b,
    float* gi, float* gh, unsigned short* tb, unsigned short* outb,
    float* __restrict__ dout, unsigned* bar) {

  const int lane  = threadIdx.x & 31;
  const int wid   = blockIdx.x * (PBLOCK >> 5) + (threadIdx.x >> 5);
  const int gtid  = blockIdx.x * PBLOCK + threadIdx.x;
  const int gsz   = PGRID * PBLOCK;
  const int mrow  = lane & 15;          // output column / A row owned by this lane
  const int mbase = (lane >> 4) << 3;   // C/D row base for this lane half
  unsigned* cnt = bar;
  unsigned* gen = bar + 1;

  const unsigned short* h0row = h0b + (size_t)mrow * KH;
  const unsigned short* h1row = h1b + (size_t)mrow * KH;
  const unsigned short* trow  = tb  + (size_t)mrow * KH;
  const unsigned short* orow  = outb + (size_t)mrow * KE;

  for (int s = 0; s < KS; ++s) {
    // ---- phase A: layer-0 gate pre-activations gi = x@Wih0.T, gh = h0@Whh0.T
    {
      const int tok = trg[mrow * KS + s];
      const unsigned short* xrow = embbf + (size_t)tok * KE;
      for (int t = wid; t < 2 * (K3 / 16); t += NWAVES) {
        const bool hpart = t >= (K3 / 16);
        const int n0 = (hpart ? t - (K3 / 16) : t) << 4;
        v8f acc = hpart ? tile_gemm_bf16(h0row, whh0, KH, n0, KH, lane)
                        : tile_gemm_bf16(xrow,  wih0, KE, n0, KE, lane);
        float* dst = hpart ? gh : gi;
        const int col = n0 + mrow;
#pragma unroll
        for (int r = 0; r < 8; ++r) dst[(mbase + r) * K3 + col] = acc[r];
      }
    }
    grid_sync(cnt, gen);

    // ---- phase B: GRU layer-0 elementwise update
    for (int i = gtid; i < KB * KH; i += gsz) {
      const int b = i >> 10, j = i & (KH - 1);
      const float* gib = gi + b * K3;
      const float* ghb = gh + b * K3;
      float r  = sigm(gib[j] + bih0[j] + ghb[j] + bhh0[j]);
      float z  = sigm(gib[KH + j] + bih0[KH + j] + ghb[KH + j] + bhh0[KH + j]);
      float nn = tanhf(gib[2 * KH + j] + bih0[2 * KH + j] +
                       r * (ghb[2 * KH + j] + bhh0[2 * KH + j]));
      float h = (1.f - z) * nn + z * h0f[i];
      h0f[i] = h; h0b[i] = f2bf(h);
    }
    grid_sync(cnt, gen);

    // ---- phase C: layer-1 gate pre-activations
    for (int t = wid; t < 2 * (K3 / 16); t += NWAVES) {
      const bool hpart = t >= (K3 / 16);
      const int n0 = (hpart ? t - (K3 / 16) : t) << 4;
      v8f acc = hpart ? tile_gemm_bf16(h1row, whh1, KH, n0, KH, lane)
                      : tile_gemm_bf16(h0row, wih1, KH, n0, KH, lane);
      float* dst = hpart ? gh : gi;
      const int col = n0 + mrow;
#pragma unroll
      for (int r = 0; r < 8; ++r) dst[(mbase + r) * K3 + col] = acc[r];
    }
    grid_sync(cnt, gen);

    // ---- phase D: GRU layer-1 elementwise update
    for (int i = gtid; i < KB * KH; i += gsz) {
      const int b = i >> 10, j = i & (KH - 1);
      const float* gib = gi + b * K3;
      const float* ghb = gh + b * K3;
      float r  = sigm(gib[j] + bih1[j] + ghb[j] + bhh1[j]);
      float z  = sigm(gib[KH + j] + bih1[KH + j] + ghb[KH + j] + bhh1[KH + j]);
      float nn = tanhf(gib[2 * KH + j] + bih1[2 * KH + j] +
                       r * (ghb[2 * KH + j] + bhh1[2 * KH + j]));
      float h = (1.f - z) * nn + z * h1f[i];
      h1f[i] = h; h1b[i] = f2bf(h);
    }
    grid_sync(cnt, gen);

    // ---- phase E: t = relu(h1 @ W1.T + b1)   (16 x 1024, bf16 out)
    for (int t = wid; t < KH / 16; t += NWAVES) {
      const int n0 = t << 4;
      v8f acc = tile_gemm_bf16(h1row, w1b, KH, n0, KH, lane);
      const int col = n0 + mrow;
      const float bb = b1[col];
#pragma unroll
      for (int r = 0; r < 8; ++r)
        tb[(mbase + r) * KH + col] = f2bf(fmaxf(acc[r] + bb, 0.f));
    }
    grid_sync(cnt, gen);

    // ---- phase F: out = t @ W2.T + b2        (16 x 512, bf16 out)
    for (int t = wid; t < KE / 16; t += NWAVES) {
      const int n0 = t << 4;
      v8f acc = tile_gemm_bf16(trow, w2b, KH, n0, KH, lane);
      const int col = n0 + mrow;
      const float bb = b2[col];
#pragma unroll
      for (int r = 0; r < 8; ++r)
        outb[(mbase + r) * KE + col] = f2bf(acc[r] + bb);
    }
    grid_sync(cnt, gen);

    // ---- phase G: logits = out @ emb.T + b_gen  (16 x 32000) -> d_out[b,s,v]
    for (int t = wid; t < KV / 16; t += NWAVES) {
      const int n0 = t << 4;
      v8f acc = tile_gemm_bf16(orow, embbf, KE, n0, KE, lane);
      const int col = n0 + mrow;
      const float bg = bgen[col];
#pragma unroll
      for (int r = 0; r < 8; ++r)
        dout[((size_t)(mbase + r) * KS + s) * KV + col] = acc[r] + bg;
    }
    // no barrier needed: phase-G buffers untouched until 5 barriers into s+1
  }
}

// ---------------- host launcher ----------------

extern "C" void kernel_launch(void* const* d_in, const int* in_sizes, int n_in,
                              void* d_out, int out_size, void* d_ws, size_t ws_size,
                              hipStream_t stream) {
  const float* hidden = (const float*)d_in[0];
  const int*   trg    = (const int*)d_in[1];
  const float* emb    = (const float*)d_in[2];
  const float* wih0   = (const float*)d_in[3];
  const float* whh0   = (const float*)d_in[4];
  const float* bih0   = (const float*)d_in[5];
  const float* bhh0   = (const float*)d_in[6];
  const float* wih1   = (const float*)d_in[7];
  const float* whh1   = (const float*)d_in[8];
  const float* bih1   = (const float*)d_in[9];
  const float* bhh1   = (const float*)d_in[10];
  const float* w1     = (const float*)d_in[11];
  const float* b1     = (const float*)d_in[12];
  const float* w2     = (const float*)d_in[13];
  const float* b2     = (const float*)d_in[14];
  const float* bgen   = (const float*)d_in[15];
  float* dout = (float*)d_out;

  char* ws = (char*)d_ws;
  size_t off = 0;
  auto alloc = [&](size_t bytes) -> void* {
    void* p = ws + off;
    off += (bytes + 255) & ~(size_t)255;
    return p;
  };

  unsigned*       bar    = (unsigned*)alloc(256);
  unsigned short* embbf  = (unsigned short*)alloc((size_t)KV * KE * 2);
  unsigned short* wih0b  = (unsigned short*)alloc((size_t)K3 * KE * 2);
  unsigned short* whh0b  = (unsigned short*)alloc((size_t)K3 * KH * 2);
  unsigned short* wih1b  = (unsigned short*)alloc((size_t)K3 * KH * 2);
  unsigned short* whh1b  = (unsigned short*)alloc((size_t)K3 * KH * 2);
  unsigned short* w1b    = (unsigned short*)alloc((size_t)KH * KH * 2);
  unsigned short* w2b    = (unsigned short*)alloc((size_t)KE * KH * 2);
  float*          h0f    = (float*)alloc((size_t)KB * KH * 4);
  float*          h1f    = (float*)alloc((size_t)KB * KH * 4);
  unsigned short* h0b    = (unsigned short*)alloc((size_t)KB * KH * 2);
  unsigned short* h1b    = (unsigned short*)alloc((size_t)KB * KH * 2);
  float*          gi     = (float*)alloc((size_t)KB * K3 * 4);
  float*          gh     = (float*)alloc((size_t)KB * K3 * 4);
  unsigned short* tb     = (unsigned short*)alloc((size_t)KB * KH * 2);
  unsigned short* outb   = (unsigned short*)alloc((size_t)KB * KE * 2);
  (void)ws_size; (void)in_sizes; (void)n_in; (void)out_size;

  // one-time fp32 -> bf16 conversions (L2-resident afterwards)
  cvt_f32_bf16<<<2048, 256, 0, stream>>>(emb,  embbf, (size_t)KV * KE);
  cvt_f32_bf16<<<512,  256, 0, stream>>>(wih0, wih0b, (size_t)K3 * KE);
  cvt_f32_bf16<<<512,  256, 0, stream>>>(whh0, whh0b, (size_t)K3 * KH);
  cvt_f32_bf16<<<512,  256, 0, stream>>>(wih1, wih1b, (size_t)K3 * KH);
  cvt_f32_bf16<<<512,  256, 0, stream>>>(whh1, whh1b, (size_t)K3 * KH);
  cvt_f32_bf16<<<256,  256, 0, stream>>>(w1,   w1b,   (size_t)KH * KH);
  cvt_f32_bf16<<<256,  256, 0, stream>>>(w2,   w2b,   (size_t)KE * KH);
  init_state<<<64, 256, 0, stream>>>(hidden, h0f, h0b, h1f, h1b, bar);

  decoder_persistent<<<PGRID, PBLOCK, 0, stream>>>(
      trg, embbf, wih0b, whh0b, bih0, bhh0, wih1b, whh1b, bih1, bhh1,
      w1b, b1, w2b, b2, bgen,
      h0f, h0b, h1f, h1b, gi, gh, tb, outb, dout, bar);
}
